// GNNSAGETransformer_5068061409375
// MI455X (gfx1250) — matrix operand order repaired and checked by reading, not compiled, not run on hardware
//
#include <hip/hip_runtime.h>

typedef __attribute__((ext_vector_type(16))) _Float16 v16h;
typedef __attribute__((ext_vector_type(8)))  float    v8f;

#define GCOUNT 64
#define NNODES 1024
#define NEDGES 16384
#define HDIM   256
#define FIN    128
#define FFDIM  1024
#define NHEADS 8
#define HD     32
#define NTOK   64   // B*T
#define LEPS   1e-5f
#define KC     64   // K-chunk staged in LDS
#define BN     64   // N columns per block

__device__ __forceinline__ int imin(int a, int b) { return a < b ? a : b; }

// ---- gfx1250 async global->LDS copy (probe-verified on this toolchain) -----
#if defined(__has_builtin)
#if __has_builtin(__builtin_amdgcn_global_load_async_to_lds_b32) && \
    __has_builtin(__builtin_amdgcn_s_wait_asynccnt)
#define USE_ASYNC_LDS 1
#endif
#endif

#ifdef USE_ASYNC_LDS
#define ASYNC_CP(ldst, gsrc)                                              \
  __builtin_amdgcn_global_load_async_to_lds_b32(                          \
      (__attribute__((address_space(1))) int*)(gsrc),                     \
      (__attribute__((address_space(3))) int*)(ldst), 0, 0)
#define ASYNC_WAIT() __builtin_amdgcn_s_wait_asynccnt(0)
#else
#define ASYNC_CP(ldst, gsrc) (*(ldst) = *(gsrc))
#define ASYNC_WAIT() ((void)0)
#endif

// ---------------------------------------------------------------------------
// 0) fp32 -> fp16 converter (for the x self-path mirror)
// ---------------------------------------------------------------------------
__global__ __launch_bounds__(256) void cvt_f16(const float* __restrict__ in,
                                               _Float16* __restrict__ out, int n) {
  int i = blockIdx.x * 256 + threadIdx.x;
  if (i < n) out[i] = (_Float16)in[i];
}

// ---------------------------------------------------------------------------
// 1) SAGE mean aggregation: per (graph, 8-feature chunk) block. LDS scatter-add
//    in fp32, result written directly as fp16 (GEMM A-operand precision).
// ---------------------------------------------------------------------------
__global__ __launch_bounds__(256) void sage_aggregate(
    const float* __restrict__ X, const int* __restrict__ EI,
    _Float16* __restrict__ OUT, int F, int chunks) {
  int g  = blockIdx.x / chunks;
  int c  = blockIdx.x % chunks;
  int f0 = c * 8;
  __shared__ float s_agg[NNODES * 8];   // 32 KB
  __shared__ float s_deg[NNODES];       //  4 KB
  int tid = threadIdx.x;
  for (int i = tid; i < NNODES * 8; i += 256) s_agg[i] = 0.f;
  for (int i = tid; i < NNODES; i += 256) s_deg[i] = 0.f;
  __syncthreads();
  const int* src = EI + (size_t)g * 2 * NEDGES;
  const int* dst = src + NEDGES;
  const float* Xg = X + (size_t)g * NNODES * F;
  for (int e = tid; e < NEDGES; e += 256) {
    int s = src[e], d = dst[e];
    atomicAdd(&s_deg[d], 1.0f);
    const float* xs = Xg + (size_t)s * F + f0;
    float* ag = s_agg + d * 8;
#pragma unroll
    for (int j = 0; j < 8; ++j) atomicAdd(&ag[j], xs[j]);
  }
  __syncthreads();
  _Float16* Og = OUT + (size_t)g * NNODES * F;
  for (int i = tid; i < NNODES * 8; i += 256) {
    int n = i >> 3, j = i & 7;
    float dg = fmaxf(s_deg[n], 1.0f);
    Og[(size_t)n * F + f0 + j] = (_Float16)(s_agg[n * 8 + j] / dg);
  }
}

// ---------------------------------------------------------------------------
// 2) WMMA GEMM: block = 128(M) x 64(N); 8 waves, each wave = 16x64 strip.
//    Weights staged K-chunk-wise in LDS via async copies. A operand type
//    templated (f16 for the big SAGE GEMMs, f32 for the small transformer).
// ---------------------------------------------------------------------------
template <typename AT>
__device__ __forceinline__ v16h load_a_frag(const AT* __restrict__ arow, int koffA) {
  v16h a;
#pragma unroll
  for (int e = 0; e < 16; e += 2) {
    int k = ((e < 8) ? e : e + 8) + koffA;  // ISA 16-bit A 16x32 layout
    a[e]     = (_Float16)arow[k];
    a[e + 1] = (_Float16)arow[k + 1];
  }
  return a;
}

template <bool MASKED>
__device__ __forceinline__ void mma_groups(const float* __restrict__ sBp,
                                           int kb, int kbB, int llow,
                                           int n0, int N, const v16h& a, v8f* acc) {
#pragma unroll
  for (int g = 0; g < 4; ++g) {
    int cloc = g * 16 + llow;
    v16h b;
    if (MASKED) {
      float vmask = ((n0 + cloc) < N) ? 1.f : 0.f;  // branch-free col guard
#pragma unroll
      for (int e = 0; e < 16; ++e)
        b[e] = (_Float16)(sBp[(kb + kbB + ((e < 8) ? e : e + 8)) * BN + cloc] * vmask);
    } else {
#pragma unroll
      for (int e = 0; e < 16; ++e)
        b[e] = (_Float16)sBp[(kb + kbB + ((e < 8) ? e : e + 8)) * BN + cloc];
    }
    acc[g] = __builtin_amdgcn_wmma_f32_16x16x32_f16(
        false, a, false, b, (short)0, acc[g], false, false);
  }
}

template <typename AT>
__global__ __launch_bounds__(256) void wmma_gemm(
    const AT* __restrict__ A, const float* __restrict__ W,
    const AT* __restrict__ A2, const float* __restrict__ W2,
    const float* __restrict__ bias, const float* __restrict__ resid,
    float* __restrict__ C, _Float16* __restrict__ Ch,
    int M, int K, int N, int act) {
  __shared__ float sB[2 * KC * BN];  // 32 KB: [k][n] chunk, second half for W2
  const int nt   = (N + BN - 1) / BN;
  const int mb   = blockIdx.x / nt;
  const int tn   = blockIdx.x % nt;
  const int n0   = tn * BN;
  const int tid  = threadIdx.x;
  const int wave = tid >> 5;
  const int lane = tid & 31;
  const int llow = lane & 15;
  const int row  = mb * 128 + wave * 16 + llow;
  const int rowC = imin(row, M - 1);       // clamp; OOB rows dropped at store
  const int koffA = (lane < 16) ? 0 : 8;   // ISA 16-bit A 16x32 layout
  const int kbB   = (lane < 16) ? 0 : 8;   // ISA 16-bit B 32x16 layout
  const bool dual = (A2 != nullptr);
  const bool fullTile = (n0 + BN <= N);    // uniform: skip masking in hot path
  float* sB2 = sB + KC * BN;

  v8f acc[4];
#pragma unroll
  for (int g = 0; g < 4; ++g) acc[g] = (v8f){};

  for (int kb0 = 0; kb0 < K; kb0 += KC) {
    // --- stage weight chunk(s) into LDS (async, coalesced along N)
    for (int idx = tid; idx < KC * BN; idx += 256) {
      int kk = idx >> 6;              // / BN
      int cc = idx & (BN - 1);
      int gc = imin(n0 + cc, N - 1);  // clamp; masked at consumption
      const float* g1 = W + (size_t)(kb0 + kk) * N + gc;
      ASYNC_CP(&sB[idx], g1);
      if (dual) {
        const float* g2 = W2 + (size_t)(kb0 + kk) * N + gc;
        ASYNC_CP(&sB2[idx], g2);
      }
    }
    ASYNC_WAIT();
    __syncthreads();

    for (int kb = 0; kb < KC; kb += 32) {
      const AT* arow = A + (size_t)rowC * K + kb0 + kb;
      __builtin_prefetch(arow + KC, 0, 1);
      v16h a = load_a_frag(arow, koffA);
      if (fullTile) mma_groups<false>(sB, kb, kbB, llow, n0, N, a, acc);
      else          mma_groups<true >(sB, kb, kbB, llow, n0, N, a, acc);
      if (dual) {
        v16h a2 = load_a_frag(A2 + (size_t)rowC * K + kb0 + kb, koffA);
        if (fullTile) mma_groups<false>(sB2, kb, kbB, llow, n0, N, a2, acc);
        else          mma_groups<true >(sB2, kb, kbB, llow, n0, N, a2, acc);
      }
    }
    __syncthreads();
  }

  // --- epilogue: C/D layout: VGPR r -> M = base + r (+8 for upper lane half)
  const int mbase = mb * 128 + wave * 16 + ((lane < 16) ? 0 : 8);
#pragma unroll
  for (int g = 0; g < 4; ++g) {
    int n = n0 + g * 16 + llow;
    if (n < N) {
      float bv = bias ? bias[n] : 0.f;
#pragma unroll
      for (int r = 0; r < 8; ++r) {
        int m = mbase + r;
        if (m < M) {
          float v = acc[g][r] + bv;
          if (resid) v += resid[(size_t)m * N + n];
          if (act) v = fmaxf(v, 0.f);
          C[(size_t)m * N + n] = v;
          if (Ch) Ch[(size_t)m * N + n] = (_Float16)v;   // f16 mirror for next GEMM
        }
      }
    }
  }
}

// ---------------------------------------------------------------------------
// 3) mean-pool over nodes: emb[g,h] = mean_n H[g,n,h]
// ---------------------------------------------------------------------------
__global__ __launch_bounds__(256) void mean_pool(const float* __restrict__ Hin,
                                                 float* __restrict__ Out) {
  int g = blockIdx.x, i = threadIdx.x;
  const float* base = Hin + (size_t)g * NNODES * HDIM + i;
  float s = 0.f;
  for (int n = 0; n < NNODES; ++n) s += base[(size_t)n * HDIM];
  Out[g * HDIM + i] = s * (1.0f / (float)NNODES);
}

// ---------------------------------------------------------------------------
// 4) attention: one block per (batch, head). T=16, HD=32 fits in LDS.
// ---------------------------------------------------------------------------
__global__ __launch_bounds__(256) void attention(const float* __restrict__ qkv,
                                                 float* __restrict__ ctx) {
  int b = blockIdx.x / NHEADS;
  int h = blockIdx.x % NHEADS;
  __shared__ float sq[16][HD], sk[16][HD], sv[16][HD], sc[16][16];
  int tid = threadIdx.x;
  for (int idx = tid; idx < 16 * HD; idx += 256) {
    int t = idx / HD, d = idx % HD;
    const float* rowp = qkv + (size_t)(b * 16 + t) * (3 * HDIM) + h * HD;
    sq[t][d] = rowp[d];
    sk[t][d] = rowp[HDIM + d];
    sv[t][d] = rowp[2 * HDIM + d];
  }
  __syncthreads();
  int i = tid >> 4, j = tid & 15;
  float s = 0.f;
#pragma unroll
  for (int d = 0; d < HD; ++d) s += sq[i][d] * sk[j][d];
  sc[i][j] = s * 0.17677669529663687f;  // 1/sqrt(32)
  __syncthreads();
  if (tid < 16) {
    float mx = sc[tid][0];
#pragma unroll
    for (int c = 1; c < 16; ++c) mx = fmaxf(mx, sc[tid][c]);
    float sum = 0.f;
#pragma unroll
    for (int c = 0; c < 16; ++c) { float e = __expf(sc[tid][c] - mx); sc[tid][c] = e; sum += e; }
    float inv = 1.0f / sum;
#pragma unroll
    for (int c = 0; c < 16; ++c) sc[tid][c] *= inv;
  }
  __syncthreads();
  for (int idx = tid; idx < 16 * HD; idx += 256) {
    int t = idx / HD, d = idx % HD;
    float a = 0.f;
#pragma unroll
    for (int c = 0; c < 16; ++c) a += sc[t][c] * sv[c][d];
    ctx[(size_t)(b * 16 + t) * HDIM + h * HD + d] = a;
  }
}

// ---------------------------------------------------------------------------
// 5) out = LayerNorm(x + r) * s + b   (one block per token, H=256)
// ---------------------------------------------------------------------------
__global__ __launch_bounds__(256) void add_layernorm(
    const float* __restrict__ x, const float* __restrict__ r,
    const float* __restrict__ sc, const float* __restrict__ bs,
    float* __restrict__ out) {
  int t = blockIdx.x, i = threadIdx.x;
  float v = x[t * HDIM + i] + r[t * HDIM + i];
  float s1 = v, s2 = v * v;
#pragma unroll
  for (int o = 16; o > 0; o >>= 1) {
    s1 += __shfl_down(s1, o, 32);
    s2 += __shfl_down(s2, o, 32);
  }
  __shared__ float red1[8], red2[8], mv[2];
  int wid = i >> 5, lane = i & 31;
  if (lane == 0) { red1[wid] = s1; red2[wid] = s2; }
  __syncthreads();
  if (i == 0) {
    float a = 0.f, b2 = 0.f;
#pragma unroll
    for (int w = 0; w < 8; ++w) { a += red1[w]; b2 += red2[w]; }
    float mean = a * (1.0f / HDIM);
    float var  = b2 * (1.0f / HDIM) - mean * mean;
    mv[0] = mean; mv[1] = rsqrtf(var + LEPS);
  }
  __syncthreads();
  out[t * HDIM + i] = (v - mv[0]) * mv[1] * sc[i] + bs[i];
}

// ---------------------------------------------------------------------------
// 6) gather last token per batch: Out[b,:] = H[b*16+15,:]
// ---------------------------------------------------------------------------
__global__ __launch_bounds__(256) void gather_last(const float* __restrict__ Hin,
                                                   float* __restrict__ Out) {
  int idx = blockIdx.x * blockDim.x + threadIdx.x;
  if (idx < 4 * HDIM) {
    int bb = idx / HDIM, j = idx % HDIM;
    Out[idx] = Hin[(size_t)(bb * 16 + 15) * HDIM + j];
  }
}

// ---------------------------------------------------------------------------
template <typename AT>
static inline void launch_gemm(hipStream_t st, const AT* A, const float* W,
                               const AT* A2, const float* W2,
                               const float* bias, const float* resid,
                               float* C, _Float16* Ch, int M, int K, int N, int act) {
  int blocks = ((M + 127) / 128) * ((N + BN - 1) / BN);
  hipLaunchKernelGGL((wmma_gemm<AT>), dim3(blocks), dim3(256), 0, st,
                     A, W, A2, W2, bias, resid, C, Ch, M, K, N, act);
}

extern "C" void kernel_launch(void* const* d_in, const int* in_sizes, int n_in,
                              void* d_out, int out_size, void* d_ws, size_t ws_size,
                              hipStream_t stream) {
  const float* x     = (const float*)d_in[0];   // [64,1024,128]
  const int*   ei    = (const int*)d_in[1];     // [64,2,16384]
  const float* s1Wl  = (const float*)d_in[2];
  const float* s1bl  = (const float*)d_in[3];
  const float* s1Wr  = (const float*)d_in[4];
  const float* s2Wl  = (const float*)d_in[5];
  const float* s2bl  = (const float*)d_in[6];
  const float* s2Wr  = (const float*)d_in[7];
  const float* Wqkv  = (const float*)d_in[8];   // [2,256,768]
  const float* bqkv  = (const float*)d_in[9];
  const float* Wo    = (const float*)d_in[10];  // [2,256,256]
  const float* bo    = (const float*)d_in[11];
  const float* ln1s  = (const float*)d_in[12];
  const float* ln1b  = (const float*)d_in[13];
  const float* W1    = (const float*)d_in[14];  // [2,256,1024]
  const float* b1    = (const float*)d_in[15];
  const float* W2    = (const float*)d_in[16];  // [2,1024,256]
  const float* b2    = (const float*)d_in[17];
  const float* ln2s  = (const float*)d_in[18];
  const float* ln2b  = (const float*)d_in[19];
  const float* fcW   = (const float*)d_in[20];  // [256,10]
  const float* fcb   = (const float*)d_in[21];

  const int M = GCOUNT * NNODES;  // 65536 rows

  // --- workspace partition (256B aligned slices)
  char* wsb = (char*)d_ws;
  auto carve = [&](size_t bytes) {
    char* p = wsb;
    wsb += (bytes + 255) & ~(size_t)255;
    return p;
  };
  _Float16* xh    = (_Float16*)carve((size_t)M * FIN  * 2);  // f16 mirror of x
  _Float16* agg1h = (_Float16*)carve((size_t)M * FIN  * 2);
  float*    h1    = (float*)   carve((size_t)M * HDIM * 4);
  _Float16* h1h   = (_Float16*)carve((size_t)M * HDIM * 2);  // f16 mirror of h1
  _Float16* agg2h = (_Float16*)carve((size_t)M * HDIM * 2);
  float*    h2    = (float*)   carve((size_t)M * HDIM * 4);
  float*    emb   = (float*)   carve((size_t)NTOK * HDIM * 4);
  float*    qkvb  = (float*)   carve((size_t)NTOK * 3 * HDIM * 4);
  float*    ctxb  = (float*)   carve((size_t)NTOK * HDIM * 4);
  float*    projb = (float*)   carve((size_t)NTOK * HDIM * 4);
  float*    ffnb  = (float*)   carve((size_t)NTOK * FFDIM * 4);
  float*    lastb = (float*)   carve((size_t)4 * HDIM * 4);

  // --- f16 mirror of x for the SAGE-1 self path
  {
    int n = M * FIN;
    hipLaunchKernelGGL(cvt_f16, dim3((n + 255) / 256), dim3(256), 0, stream, x, xh, n);
  }

  // --- GraphSAGE layer 1: mean-agg (f16 out) + fused dual WMMA GEMM + relu
  hipLaunchKernelGGL(sage_aggregate, dim3(GCOUNT * (FIN / 8)), dim3(256), 0, stream,
                     x, ei, agg1h, FIN, FIN / 8);
  launch_gemm<_Float16>(stream, agg1h, s1Wl, xh, s1Wr, s1bl, nullptr,
                        h1, h1h, M, FIN, HDIM, /*relu=*/1);

  // --- GraphSAGE layer 2
  hipLaunchKernelGGL(sage_aggregate, dim3(GCOUNT * (HDIM / 8)), dim3(256), 0, stream,
                     h1, ei, agg2h, HDIM, HDIM / 8);
  launch_gemm<_Float16>(stream, agg2h, s2Wl, h1h, s2Wr, s2bl, nullptr,
                        h2, (_Float16*)nullptr, M, HDIM, HDIM, /*relu=*/0);

  // --- mean pool over nodes -> emb [64, 256]
  hipLaunchKernelGGL(mean_pool, dim3(GCOUNT), dim3(256), 0, stream, h2, emb);

  // --- Transformer encoder, L=2 (post-norm, relu FFN); small fp32 GEMMs
  for (int l = 0; l < 2; ++l) {
    const float* Wqkv_l = Wqkv + (size_t)l * HDIM * 3 * HDIM;
    const float* bqkv_l = bqkv + (size_t)l * 3 * HDIM;
    const float* Wo_l   = Wo   + (size_t)l * HDIM * HDIM;
    const float* bo_l   = bo   + (size_t)l * HDIM;
    const float* W1_l   = W1   + (size_t)l * HDIM * FFDIM;
    const float* b1_l   = b1   + (size_t)l * FFDIM;
    const float* W2_l   = W2   + (size_t)l * FFDIM * HDIM;
    const float* b2_l   = b2   + (size_t)l * HDIM;

    launch_gemm<float>(stream, emb, Wqkv_l, (const float*)nullptr, nullptr,
                       bqkv_l, nullptr, qkvb, (_Float16*)nullptr,
                       NTOK, HDIM, 3 * HDIM, 0);
    hipLaunchKernelGGL(attention, dim3(4 * NHEADS), dim3(256), 0, stream, qkvb, ctxb);
    launch_gemm<float>(stream, ctxb, Wo_l, (const float*)nullptr, nullptr,
                       bo_l, nullptr, projb, (_Float16*)nullptr,
                       NTOK, HDIM, HDIM, 0);
    hipLaunchKernelGGL(add_layernorm, dim3(NTOK), dim3(256), 0, stream,
                       emb, projb, ln1s + (size_t)l * HDIM, ln1b + (size_t)l * HDIM, emb);
    launch_gemm<float>(stream, emb, W1_l, (const float*)nullptr, nullptr,
                       b1_l, nullptr, ffnb, (_Float16*)nullptr,
                       NTOK, HDIM, FFDIM, /*relu=*/1);
    launch_gemm<float>(stream, ffnb, W2_l, (const float*)nullptr, nullptr,
                       b2_l, nullptr, projb, (_Float16*)nullptr,
                       NTOK, FFDIM, HDIM, 0);
    hipLaunchKernelGGL(add_layernorm, dim3(NTOK), dim3(256), 0, stream,
                       emb, projb, ln2s + (size_t)l * HDIM, ln2b + (size_t)l * HDIM, emb);
  }

  // --- classifier on last token per batch: [4,256] x [256,10] -> d_out [4,10]
  hipLaunchKernelGGL(gather_last, dim3(4), dim3(256), 0, stream, emb, lastb);
  launch_gemm<float>(stream, lastb, fcW, (const float*)nullptr, nullptr,
                     fcb, nullptr, (float*)d_out, (_Float16*)nullptr,
                     4, HDIM, 10, 0);
}
